// MultiHeadAttention_27968827031948
// MI455X (gfx1250) — matrix operand order repaired and checked by reading, not compiled, not run on hardware
//
#include <hip/hip_runtime.h>
#include <hip/hip_bf16.h>

// ---- MHA: B=2, S=2048, D=1024, H=16, Dh=64 ----
// d_out = out[2,2048,1024] (f32) ++ attn[2,16,2048,2048] (f32)
//
// Roofline: ~69 GFLOP total, but attn materialization dominates memory.
// Fused scores->softmax->ctx writes attn exactly once (537 MB, ~23 us at
// 23.3 TB/s) instead of a 4-pass staged version (~2.1 GB).

typedef __attribute__((ext_vector_type(16))) _Float16 v16h;
typedef __attribute__((ext_vector_type(8)))  _Float16 v8h;
typedef __attribute__((ext_vector_type(8)))  float    v8f;
typedef __attribute__((ext_vector_type(4)))  float    v4f;

__device__ __forceinline__ v8f wmma_f16(v16h a, v16h b, v8f c) {
  // D = A(16x32 f16) * B(32x16 f16) + C(16x16 f32)
  return __builtin_amdgcn_wmma_f32_16x16x32_f16(
      /*neg_a=*/false, a, /*neg_b=*/false, b,
      /*c_mod=*/(short)0, c, /*reuse_a=*/false, /*reuse_b=*/false);
}

// A fragment (16x32, 16-bit): lane<16 -> K kb..kb+15, kb=k0;
// lane>=16 -> kb=k0+16. Caller passes kb = k0 + hi*16.
__device__ __forceinline__ v16h a_frag_f16(const _Float16* __restrict__ row, int kb) {
  v8h x = *(const v8h*)(row + kb);
  v8h y = *(const v8h*)(row + kb + 8);
  v16h r;
#pragma unroll
  for (int i = 0; i < 8; ++i) { r[i] = x[i]; r[8 + i] = y[i]; }
  return r;
}

__device__ __forceinline__ v16h a_frag_f32(const float* __restrict__ row, int kb) {
  v4f x0 = *(const v4f*)(row + kb);
  v4f x1 = *(const v4f*)(row + kb + 4);
  v4f y0 = *(const v4f*)(row + kb + 8);
  v4f y1 = *(const v4f*)(row + kb + 12);
  v16h r;
#pragma unroll
  for (int i = 0; i < 4; ++i) {
    r[i]      = (_Float16)x0[i];
    r[4 + i]  = (_Float16)x1[i];
    r[8 + i]  = (_Float16)y0[i];
    r[12 + i] = (_Float16)y1[i];
  }
  return r;
}

// B fragment (32x16, 16-bit): lane<16 holds K=k0..k0+15 of column n, lane>=16
// holds K=k0+16..k0+31. Caller passes kb = k0 + hi*16. Column n is supplied as
// a contiguous "B-transposed" row.
__device__ __forceinline__ v16h b_frag_f16(const _Float16* __restrict__ row, int kb) {
  v8h x = *(const v8h*)(row + kb);
  v8h y = *(const v8h*)(row + kb + 8);
  v16h r;
#pragma unroll
  for (int i = 0; i < 8; ++i) { r[i] = x[i]; r[8 + i] = y[i]; }
  return r;
}

__device__ __forceinline__ v16h b_frag_f32(const float* __restrict__ row, int kb) {
  v4f x0 = *(const v4f*)(row + kb);
  v4f x1 = *(const v4f*)(row + kb + 4);
  v4f x2 = *(const v4f*)(row + kb + 8);
  v4f x3 = *(const v4f*)(row + kb + 12);
  v16h r;
#pragma unroll
  for (int i = 0; i < 4; ++i) {
    r[i]      = (_Float16)x0[i];
    r[4 + i]  = (_Float16)x1[i];
    r[8 + i]  = (_Float16)x2[i];
    r[12 + i] = (_Float16)x3[i];
  }
  return r;
}

// ---------------------------------------------------------------------------
// Projection: P = X @ W^T, X [4096,1024] f32, W [1024,1024] f32.
// Wave computes a 16x64 strip. Output f16, head-split:
//   transposeV==0 : Out[b,h,s,dh]  (Q, K)   transposeV==1 : Out[b,h,dh,s] (V)
__global__ void __launch_bounds__(256) proj_kernel(
    const float* __restrict__ X, const float* __restrict__ W,
    _Float16* __restrict__ Out, int transposeV) {
  const int lane = threadIdx.x & 31;
  const int lo = lane & 15;
  const int hi = lane >> 4;
  const int wave = blockIdx.x * 8 + (threadIdx.x >> 5);
  const int m0 = (wave & 255) * 16;
  const int n0 = (wave >> 8) * 64;
  const float* arow = X + (size_t)(m0 + lo) * 1024;
  const float* brow = W + (size_t)(n0 + lo) * 1024;
  const int ka = hi * 16, kbo = hi * 16;
  v8f acc[4] = {};
  for (int k0 = 0; k0 < 1024; k0 += 32) {
    v16h a = a_frag_f32(arow, k0 + ka);
#pragma unroll
    for (int j = 0; j < 4; ++j) {
      v16h b = b_frag_f32(brow + (size_t)j * 16 * 1024, k0 + kbo);
      acc[j] = wmma_f16(a, b, acc[j]);
    }
  }
#pragma unroll
  for (int j = 0; j < 4; ++j) {
    const int n = n0 + j * 16 + lo;
    const int h = n >> 6, dh = n & 63;
#pragma unroll
    for (int r = 0; r < 8; ++r) {
      const int m = m0 + hi * 8 + r;
      const int bb = m >> 11, s = m & 2047;
      const size_t idx = transposeV
          ? ((size_t)((bb * 16 + h) * 64 + dh)) * 2048 + s
          : ((size_t)((bb * 16 + h) * 2048 + s)) * 64 + dh;
      Out[idx] = (_Float16)acc[j][r];
    }
  }
}

// ---------------------------------------------------------------------------
// Fused attention: one block (8 wave32s) per (b, h, 16-row q-tile).
//   phase 1: scores tile 16x2048 into padded LDS (each wave: 16x256 strip)
//   phase 2: softmax, 2 wave-private rows per wave, row held in registers;
//            normalized attn streamed to HBM with nontemporal stores (write-once)
//   phase 3: ctx = attn_tile @ V  (split-k across wave pairs, LDS reduction)
#define SROW 2056  // 2048 + 8-float pad: breaks bank alignment of row stride

__global__ void __launch_bounds__(256) attn_fused_kernel(
    const _Float16* __restrict__ Qh, const _Float16* __restrict__ Kh,
    const _Float16* __restrict__ Vt, float* __restrict__ Attn,
    _Float16* __restrict__ Ctx) {
  extern __shared__ char smem[];
  float*    sS   = (float*)smem;                                  // [16][SROW]
  float*    sRed = (float*)(smem + (size_t)16 * SROW * 4);        // [8][256]
  _Float16* sQ   = (_Float16*)(smem + (size_t)16 * SROW * 4 + 8 * 256 * 4); // [16][64]

  const int t = threadIdx.x;
  const int lane = t & 31, lo = lane & 15, hi = lane >> 4;
  const int wv = t >> 5;                 // wave 0..7
  const int bh = blockIdx.x >> 7;        // b*16+h
  const int q0 = (blockIdx.x & 127) * 16;

  // ---- stage shared Q tile (16x64 f16 = 2 KB) into LDS ----
  const _Float16* Qg = Qh + ((size_t)bh * 2048 + q0) * 64;
  if (t < 128) {
    const int off = t * 8;  // 16 bytes per thread
    *(v8h*)(sQ + off) = *(const v8h*)(Qg + off);
  }
  __syncthreads();

  // ---- phase 1: scores ----
  const _Float16* qrow = sQ + lo * 64;
  const v16h a0 = a_frag_f16(qrow, 0 + hi * 16);
  const v16h a1 = a_frag_f16(qrow, 32 + hi * 16);
  const _Float16* Kg = Kh + (size_t)bh * 2048 * 64;
#pragma unroll 2
  for (int nt = 0; nt < 16; ++nt) {
    const int n0 = wv * 256 + nt * 16;
    const _Float16* brow = Kg + (size_t)(n0 + lo) * 64;
    v8f acc = {};
    acc = wmma_f16(a0, b_frag_f16(brow, 0 + hi * 16), acc);
    acc = wmma_f16(a1, b_frag_f16(brow, 32 + hi * 16), acc);
#pragma unroll
    for (int r = 0; r < 8; ++r)
      sS[(size_t)(hi * 8 + r) * SROW + n0 + lo] = acc[r] * 0.125f;  // 1/sqrt(64)
  }
  __syncthreads();

  // ---- phase 2: softmax (rows 2*wv, 2*wv+1 are wave-private) ----
  float* attn_base = Attn + ((size_t)bh * 2048 + q0) * 2048;
#pragma unroll
  for (int rr = 0; rr < 2; ++rr) {
    const int row = wv * 2 + rr;
    float* srow = sS + (size_t)row * SROW;
    v4f x[16];
    float mx = -3.4e38f;
#pragma unroll
    for (int c = 0; c < 16; ++c) {
      x[c] = *(const v4f*)(srow + c * 128 + lane * 4);
#pragma unroll
      for (int e = 0; e < 4; ++e) mx = fmaxf(mx, x[c][e]);
    }
#pragma unroll
    for (int off = 16; off > 0; off >>= 1) mx = fmaxf(mx, __shfl_xor(mx, off, 32));
    float s = 0.f;
#pragma unroll
    for (int c = 0; c < 16; ++c)
#pragma unroll
      for (int e = 0; e < 4; ++e) { x[c][e] = __expf(x[c][e] - mx); s += x[c][e]; }
#pragma unroll
    for (int off = 16; off > 0; off >>= 1) s += __shfl_xor(s, off, 32);
    const float inv = 1.0f / s;
    float* grow = attn_base + (size_t)row * 2048;
#pragma unroll
    for (int c = 0; c < 16; ++c) {
      v4f y;
#pragma unroll
      for (int e = 0; e < 4; ++e) y[e] = x[c][e] * inv;
      *(v4f*)(srow + c * 128 + lane * 4) = y;  // LDS copy for ctx GEMM
      __builtin_nontemporal_store(y, (v4f*)(grow + c * 128 + lane * 4));
    }
  }
  __syncthreads();

  // ---- phase 3: ctx = attn_tile(16x2048) @ V(2048x64), split-k by 2 ----
  const int dt = wv & 3;     // d-tile (16 cols each)
  const int kh2 = wv >> 2;   // k-half
  const _Float16* vrow = Vt + ((size_t)bh * 64 + dt * 16 + lo) * 2048 + kh2 * 1024;
  const float* arow = sS + (size_t)lo * SROW + kh2 * 1024;
  v8f acc = {};
  for (int k0 = 0; k0 < 1024; k0 += 32) {
    v16h a = a_frag_f32(arow, k0 + hi * 16);    // LDS, f32 -> f16
    v16h b = b_frag_f16(vrow, k0 + hi * 16);    // global (L2-resident)
    acc = wmma_f16(a, b, acc);
  }
  float* red = sRed + wv * 256;
#pragma unroll
  for (int r = 0; r < 8; ++r) red[(hi * 8 + r) * 16 + lo] = acc[r];
  __syncthreads();
  if (wv < 4) {
    const int b_ = bh >> 4, h_ = bh & 15;
#pragma unroll
    for (int r = 0; r < 8; ++r) {
      const int m = hi * 8 + r;
      const float sum =
          sRed[wv * 256 + m * 16 + lo] + sRed[(wv + 4) * 256 + m * 16 + lo];
      Ctx[((size_t)b_ * 2048 + q0 + m) * 1024 + h_ * 64 + wv * 16 + lo] =
          (_Float16)sum;
    }
  }
}

// ---------------------------------------------------------------------------
// out = ctx @ w_fc^T + b_fc  (f32 output).
__global__ void __launch_bounds__(256) fc_kernel(
    const _Float16* __restrict__ Ctx, const float* __restrict__ Wfc,
    const float* __restrict__ bfc, float* __restrict__ Out) {
  const int lane = threadIdx.x & 31;
  const int lo = lane & 15;
  const int hi = lane >> 4;
  const int wave = blockIdx.x * 8 + (threadIdx.x >> 5);
  const int m0 = (wave & 255) * 16;
  const int n0 = (wave >> 8) * 64;
  const _Float16* arow = Ctx + (size_t)(m0 + lo) * 1024;
  const float* brow = Wfc + (size_t)(n0 + lo) * 1024;
  const int ka = hi * 16, kbo = hi * 16;
  v8f acc[4] = {};
  for (int k0 = 0; k0 < 1024; k0 += 32) {
    v16h a = a_frag_f16(arow, k0 + ka);
#pragma unroll
    for (int j = 0; j < 4; ++j) {
      v16h b = b_frag_f32(brow + (size_t)j * 16 * 1024, k0 + kbo);
      acc[j] = wmma_f16(a, b, acc[j]);
    }
  }
#pragma unroll
  for (int j = 0; j < 4; ++j) {
    const int n = n0 + j * 16 + lo;
    const float bias = bfc[n];
#pragma unroll
    for (int r = 0; r < 8; ++r) {
      const int m = m0 + hi * 8 + r;
      Out[(size_t)m * 1024 + n] = acc[j][r] + bias;
    }
  }
}

// ---------------------------------------------------------------------------
extern "C" void kernel_launch(void* const* d_in, const int* in_sizes, int n_in,
                              void* d_out, int out_size, void* d_ws, size_t ws_size,
                              hipStream_t stream) {
  const float* q    = (const float*)d_in[0];
  const float* k    = (const float*)d_in[1];
  const float* v    = (const float*)d_in[2];
  const float* w_q  = (const float*)d_in[3];
  const float* w_k  = (const float*)d_in[4];
  const float* w_v  = (const float*)d_in[5];
  const float* w_fc = (const float*)d_in[6];
  const float* b_fc = (const float*)d_in[7];

  float* out  = (float*)d_out;                       // [2,2048,1024]
  float* attn = out + (size_t)2 * 2048 * 1024;       // [2,16,2048,2048]

  // Workspace: 4 x 8 MiB f16 buffers (fully overwritten every call).
  char* ws = (char*)d_ws;
  _Float16* Qh  = (_Float16*)(ws);                        // [B,H,S,Dh]
  _Float16* Kh  = (_Float16*)(ws + (size_t)8  * 1048576); // [B,H,S,Dh]
  _Float16* Vt  = (_Float16*)(ws + (size_t)16 * 1048576); // [B,H,Dh,S]
  _Float16* Ctx = (_Float16*)(ws + (size_t)24 * 1048576); // [B,S,D]

  proj_kernel<<<512, 256, 0, stream>>>(q, w_q, Qh, 0);
  proj_kernel<<<512, 256, 0, stream>>>(k, w_k, Kh, 0);
  proj_kernel<<<512, 256, 0, stream>>>(v, w_v, Vt, 1);

  const size_t smem = (size_t)16 * SROW * 4   // score tile (padded)
                    + (size_t)8 * 256 * 4     // split-k reduction buffer
                    + (size_t)16 * 64 * 2;    // staged Q tile
  attn_fused_kernel<<<4096, 256, smem, stream>>>(Qh, Kh, Vt, attn, Ctx);

  fc_kernel<<<512, 256, 0, stream>>>(Ctx, w_fc, b_fc, out);
}
